// ImplicitGNN_52046413692943
// MI455X (gfx1250) — compile-verified, matrix operand order checked by reading, not run on hardware
//
#include <hip/hip_runtime.h>

// ---------------------------------------------------------------------------
// ImplicitGNN on MI455X (gfx1250): per-node message+update MLPs via WMMA.
// Activations carried transposed (nodes = WMMA N dim, weights = A operand) so
// the f32 C/D layout converts to the next layer's f16 B layout with per-lane
// v_cvt_pk_f16_f32 only (no LDS transpose between layers). Two 16-node tiles
// are evaluated with interleaved chains to fill WMMA hazard windows.
// ---------------------------------------------------------------------------

typedef __attribute__((ext_vector_type(16))) _Float16 v16h;
typedef __attribute__((ext_vector_type(8)))  float    v8f;

#define DEVINL __device__ __forceinline__

DEVINL v8f wmma_f16(v16h a, v16h b, v8f c) {
  // D(f32,16x16) = A(f16,16x32) * B(f16,32x16) + C
  return __builtin_amdgcn_wmma_f32_16x16x32_f16(
      /*neg_a=*/false, a, /*neg_b=*/false, b,
      /*c_mod=*/(short)0, c, /*reuse_a=*/false, /*reuse_b=*/false);
}

DEVINL v16h zero16h() {
  v16h z;
#pragma unroll
  for (int i = 0; i < 16; ++i) z[i] = (_Float16)0.f;
  return z;
}

DEVINL v8f zero8f() {
  v8f z;
#pragma unroll
  for (int i = 0; i < 8; ++i) z[i] = 0.f;
  return z;
}

DEVINL v8f relu8(v8f d) {
#pragma unroll
  for (int i = 0; i < 8; ++i) d[i] = fmaxf(d[i], 0.f);
  return d;
}

// Build an A-operand fragment holding W^T (M=out rows, K=in cols) for one
// 16-row M tile and one 32-deep K step. f16 A layout (ISA 7.12.2):
//   element e -> VGPR v=e/2, half e&1;  K = 2*(v&3) + (e&1) + 8*laneHi + 16*(v>=4)
template <int IN, int OUT>
DEVINL v16h makeA(const float* __restrict__ W, int mtile, int kstep, int lane) {
  const int hi = (lane >= 16) ? 8 : 0;
  const int m  = (mtile << 4) + (lane & 15);
  v16h a;
#pragma unroll
  for (int e = 0; e < 16; ++e) {
    const int v  = e >> 1;
    const int kk = ((v & 3) * 2 + (e & 1)) + ((v & 4) ? 16 : 0) + hi + kstep * 32;
    float w = 0.f;
    if (kk < IN && m < OUT) w = W[kk * OUT + m];  // A[m][k] = W[k][m]
    a[e] = (_Float16)w;
  }
  return a;
}

// Bias fragment in C/D layout: slot i -> row m = 16*mtile + i + 8*laneHi.
template <int OUT>
DEVINL v8f makeBias(const float* __restrict__ b, int mtile, int lane, float scale) {
  const int hi = (lane >= 16) ? 8 : 0;
  v8f r;
#pragma unroll
  for (int i = 0; i < 8; ++i) {
    const int m = (mtile << 4) + hi + i;
    r[i] = (m < OUT) ? scale * b[m] : 0.f;
  }
  return r;
}

// Convert activations in C/D layout (rows 0..31 across two f32 tiles) into an
// f16 B-operand fragment. Pure per-lane packing: B VGPR v takes D slots
// (2*(v&3), 2*(v&3)+1) of tile (v<4 ? lo : hi) — the 8*laneHi row offset of the
// C layout lines up exactly with the B layout's K offset.
DEVINL v16h packB(v8f t_lo, v8f t_hi) {
  v16h b;
#pragma unroll
  for (int v = 0; v < 8; ++v) {
    const int i = (v & 3) * 2;
    const float s0 = (v < 4) ? t_lo[i]     : t_hi[i];
    const float s1 = (v < 4) ? t_lo[i + 1] : t_hi[i + 1];
    b[2 * v]     = (_Float16)s0;
    b[2 * v + 1] = (_Float16)s1;
  }
  return b;
}

__global__ void __launch_bounds__(256)
implicit_gnn_wmma_kernel(const float* __restrict__ h,
                         const float* __restrict__ mW1, const float* __restrict__ mb1,
                         const float* __restrict__ mW2, const float* __restrict__ mb2,
                         const float* __restrict__ mW3, const float* __restrict__ mb3,
                         const float* __restrict__ mW4, const float* __restrict__ mb4,
                         const float* __restrict__ uW1, const float* __restrict__ ub1,
                         const float* __restrict__ uW2, const float* __restrict__ ub2,
                         const float* __restrict__ uW3, const float* __restrict__ ub3,
                         const float* __restrict__ uW4, const float* __restrict__ ub4,
                         float* __restrict__ out, int n) {
  const int  lane = (int)(threadIdx.x & 31);
  const bool lo   = lane < 16;

  // ---- hoisted weight fragments (loop-invariant, built once per wave) ----
  v16h mA1f[3], uA1f[3];
#pragma unroll
  for (int t = 0; t < 3; ++t) {
    mA1f[t] = makeA<2, 36>(mW1, t, 0, lane);
    uA1f[t] = makeA<4, 36>(uW1, t, 0, lane);
  }
  v16h mA2f[2] = {makeA<36, 8>(mW2, 0, 0, lane), makeA<36, 8>(mW2, 0, 1, lane)};
  v16h uA2f[2] = {makeA<36, 8>(uW2, 0, 0, lane), makeA<36, 8>(uW2, 0, 1, lane)};
  v16h mA3f = makeA<8, 3>(mW3, 0, 0, lane);
  v16h uA3f = makeA<8, 3>(uW3, 0, 0, lane);
  v16h mA4f = makeA<3, 2>(mW4, 0, 0, lane);
  v16h uA4f = makeA<3, 1>(uW4, 0, 0, lane);

  v8f mb1f[3], ub1f[3];
#pragma unroll
  for (int t = 0; t < 3; ++t) {
    mb1f[t] = makeBias<36>(mb1, t, lane, 1.f);
    ub1f[t] = makeBias<36>(ub1, t, lane, 1.f);
  }
  v8f mb2f = makeBias<8>(mb2, 0, lane, 1.f);
  v8f mb3f = makeBias<3>(mb3, 0, lane, 1.f);
  v8f mb4f = makeBias<2>(mb4, 0, lane, 4.f);  // 4 edges each add b4 -> fold 4*b4
  v8f ub2f = makeBias<8>(ub2, 0, lane, 1.f);
  v8f ub3f = makeBias<3>(ub3, 0, lane, 1.f);
  v8f ub4f = makeBias<1>(ub4, 0, lane, 1.f);

  // Evaluate two independent 16-node tiles with interleaved WMMA chains so the
  // scheduler can fill each WMMA's D->VALU hazard window with the peer chain.
  // ra/rb: 5 float2 of h columns per tile; node j of a tile lives in lane j
  // (lanes 0-15). Hi-lane B elements that pair with structurally-zero A
  // weights are left as (finite) garbage: 0 * finite == 0.
  auto evalPair = [&](const float2* ra, const float2* rb, float& oA, float& oB) {
    const v8f z8 = zero8f();
    v8f accA = mb4f, accB = mb4f;  // init with 4*b4
#pragma unroll
    for (int e = 0; e < 4; ++e) {
      // B = x^T : K=2 rows (h[2e], h[2e+1]), 16 node columns
      v16h b1A = zero16h(), b1B = zero16h();
      b1A[0] = (_Float16)ra[e].x;  b1A[1] = (_Float16)ra[e].y;
      b1B[0] = (_Float16)rb[e].x;  b1B[1] = (_Float16)rb[e].y;
      v8f d0A = wmma_f16(mA1f[0], b1A, mb1f[0]);
      v8f d0B = wmma_f16(mA1f[0], b1B, mb1f[0]);
      v8f d1A = wmma_f16(mA1f[1], b1A, mb1f[1]);
      v8f d1B = wmma_f16(mA1f[1], b1B, mb1f[1]);
      v8f d2A = wmma_f16(mA1f[2], b1A, mb1f[2]);
      v8f d2B = wmma_f16(mA1f[2], b1B, mb1f[2]);
      d0A = relu8(d0A); d0B = relu8(d0B);
      d1A = relu8(d1A); d1B = relu8(d1B);
      d2A = relu8(d2A); d2B = relu8(d2B);
      v16h b2aA = packB(d0A, d1A), b2aB = packB(d0B, d1B);  // K = 0..31
      v16h b2bA = packB(d2A, z8),  b2bB = packB(d2B, z8);   // K = 32..35
      v8f d3A = wmma_f16(mA2f[0], b2aA, mb2f);
      v8f d3B = wmma_f16(mA2f[0], b2aB, mb2f);
      d3A = wmma_f16(mA2f[1], b2bA, d3A);                   // 36 -> 8
      d3B = wmma_f16(mA2f[1], b2bB, d3B);
      d3A = relu8(d3A); d3B = relu8(d3B);
      v16h b3A = packB(d3A, z8), b3B = packB(d3B, z8);
      v8f d4A = wmma_f16(mA3f, b3A, mb3f);                  // 8 -> 3
      v8f d4B = wmma_f16(mA3f, b3B, mb3f);
      d4A = relu8(d4A); d4B = relu8(d4B);
      v16h b4A = packB(d4A, z8), b4B = packB(d4B, z8);
      accA = wmma_f16(mA4f, b4A, accA);                     // 3 -> 2, += edges
      accB = wmma_f16(mA4f, b4B, accB);
    }
    // ---- update MLP: u_in = [msum0, msum1, h_center, phi_u] (K=4) ----
    v16h u1A = zero16h(), u1B = zero16h();
    u1A[0] = (_Float16)accA[0];  u1A[1] = (_Float16)accA[1];  // hi: rows 8,9 == 0
    u1B[0] = (_Float16)accB[0];  u1B[1] = (_Float16)accB[1];
    u1A[2] = (_Float16)ra[4].x;  u1A[3] = (_Float16)ra[4].y;  // h_center, phi_u
    u1B[2] = (_Float16)rb[4].x;  u1B[3] = (_Float16)rb[4].y;
    v8f e0A = wmma_f16(uA1f[0], u1A, ub1f[0]);
    v8f e0B = wmma_f16(uA1f[0], u1B, ub1f[0]);
    v8f e1A = wmma_f16(uA1f[1], u1A, ub1f[1]);
    v8f e1B = wmma_f16(uA1f[1], u1B, ub1f[1]);
    v8f e2A = wmma_f16(uA1f[2], u1A, ub1f[2]);
    v8f e2B = wmma_f16(uA1f[2], u1B, ub1f[2]);
    e0A = relu8(e0A); e0B = relu8(e0B);
    e1A = relu8(e1A); e1B = relu8(e1B);
    e2A = relu8(e2A); e2B = relu8(e2B);
    v16h c2aA = packB(e0A, e1A), c2aB = packB(e0B, e1B);
    v16h c2bA = packB(e2A, z8),  c2bB = packB(e2B, z8);
    v8f fA = wmma_f16(uA2f[0], c2aA, ub2f);
    v8f fB = wmma_f16(uA2f[0], c2aB, ub2f);
    fA = wmma_f16(uA2f[1], c2bA, fA);                       // 36 -> 8
    fB = wmma_f16(uA2f[1], c2bB, fB);
    fA = relu8(fA); fB = relu8(fB);
    v16h c3A = packB(fA, z8), c3B = packB(fB, z8);
    v8f gA = wmma_f16(uA3f, c3A, ub3f);                     // 8 -> 3
    v8f gB = wmma_f16(uA3f, c3B, ub3f);
    gA = relu8(gA); gB = relu8(gB);
    v16h c4A = packB(gA, z8), c4B = packB(gB, z8);
    v8f oAt = wmma_f16(uA4f, c4A, ub4f);                    // 3 -> 1
    v8f oBt = wmma_f16(uA4f, c4B, ub4f);
    oA = oAt[0];  // row 0 = output, node = lane (lanes 0-15)
    oB = oBt[0];
  };

  // ---- scalar (wave-uniform) grid-stride loop over 32-node chunks ----
  // readfirstlane forces the trip count into SGPRs: no exec-mask voting, and
  // EXEC stays all-ones through every WMMA (ISA requirement).
  const int waveLocal = __builtin_amdgcn_readfirstlane((int)(threadIdx.x >> 5));
  const int wavesPerBlock = (int)(blockDim.x >> 5);
  const int waveId = (int)blockIdx.x * wavesPerBlock + waveLocal;
  const int nWaves = (int)gridDim.x * wavesPerBlock;
  const int nIter  = n >> 5;

  for (int it = waveId; it < nIter; it += nWaves) {
    const int node = (it << 5) + lane;               // lane L holds node base+L
    const float* hp = h + (size_t)node * 10;
    float2 ra[5], rb[5];
#pragma unroll
    for (int c = 0; c < 5; ++c) ra[c] = *(const float2*)(hp + 2 * c);

    if (it + nWaves < nIter)  // scalar branch; global_prefetch_b8 next chunk
      __builtin_prefetch(h + ((size_t)node + (size_t)nWaves * 32) * 10, 0, 3);

    // tile B: nodes base+16..base+31 -> swap half-waves into lanes 0-15
#pragma unroll
    for (int c = 0; c < 5; ++c) {
      rb[c].x = __shfl_xor(ra[c].x, 16, 32);
      rb[c].y = __shfl_xor(ra[c].y, 16, 32);
    }

    float outA, outB;
    evalPair(ra, rb, outA, outB);

    // single full-wave coalesced b32 store for all 32 nodes
    float outBsw = __shfl_xor(outB, 16, 32);
    out[(it << 5) + lane] = lo ? outA : outBsw;
  }
}

extern "C" void kernel_launch(void* const* d_in, const int* in_sizes, int n_in,
                              void* d_out, int out_size, void* d_ws, size_t ws_size,
                              hipStream_t stream) {
  (void)n_in; (void)d_ws; (void)ws_size; (void)out_size;
  const float* h   = (const float*)d_in[0];
  const float* mW1 = (const float*)d_in[1];  const float* mb1 = (const float*)d_in[2];
  const float* mW2 = (const float*)d_in[3];  const float* mb2 = (const float*)d_in[4];
  const float* mW3 = (const float*)d_in[5];  const float* mb3 = (const float*)d_in[6];
  const float* mW4 = (const float*)d_in[7];  const float* mb4 = (const float*)d_in[8];
  const float* uW1 = (const float*)d_in[9];  const float* ub1 = (const float*)d_in[10];
  const float* uW2 = (const float*)d_in[11]; const float* ub2 = (const float*)d_in[12];
  const float* uW3 = (const float*)d_in[13]; const float* ub3 = (const float*)d_in[14];
  const float* uW4 = (const float*)d_in[15]; const float* ub4 = (const float*)d_in[16];
  float* out = (float*)d_out;
  const int n = in_sizes[0] / 10;

  // 1024 blocks x 256 threads = 8192 waves; each wave runs ~8 chunks of 32
  // nodes for N=2^21, amortizing the per-wave weight-fragment build.
  dim3 grid(1024), block(256);
  implicit_gnn_wmma_kernel<<<grid, block, 0, stream>>>(
      h, mW1, mb1, mW2, mb2, mW3, mb3, mW4, mb4,
      uW1, ub1, uW2, ub2, uW3, ub3, uW4, ub4, out, n);
}
